// CNNModel_47785806135777
// MI455X (gfx1250) — matrix operand order repaired and checked by reading, not compiled, not run on hardware
//
#include <hip/hip_runtime.h>
#include <cstdint>
#include <cstddef>

// ---------------------------------------------------------------------------
// CDNA5 (gfx1250, wave32) implementation of the ASF-BP spiking CNN forward.
// Output value == (Tf0 @ wf1^T) / 100  (all straight-through terms cancel),
// so we only carry membranes m1,m1s,m2,m2s,mf0 and the spike-count Tf0.
// Heavy ops use v_wmma_f32_16x16x32_f16 with split-f16 weights (w=hi+lo) for
// fp32-class accuracy (spike activations are exact in f16).
// ---------------------------------------------------------------------------

typedef __attribute__((ext_vector_type(16))) _Float16     v16h;
typedef __attribute__((ext_vector_type(8)))  float        v8f;
typedef __attribute__((ext_vector_type(4)))  unsigned int u32x4;

union FragU { u32x4 u[2]; v16h v; };

// A-frag (16x32 f16): lanes 0-15: M=lane, K=[0..8)+[16..24); lanes 16-31: K=[8..16)+[24..32)
__device__ __forceinline__ v16h load_a_frag(const _Float16* row, unsigned lane) {
  unsigned kb = (lane & 16u) >> 1; // 0 or 8
  FragU f;
  f.u[0] = *(const u32x4*)(row + kb);
  f.u[1] = *(const u32x4*)(row + 16 + kb);
  return f.v;
}
// B-frag (32x16 f16), B tile stored N-major in LDS (row = column N of B):
// lanes 0-15: N=lane, K=[0..8)+[16..24); lanes 16-31: N=lane-16, K=[8..16)+[24..32)
__device__ __forceinline__ v16h load_b_frag(const _Float16* row, unsigned lane) {
  unsigned kb = (lane & 16u) >> 1; // 0 or 8
  FragU f;
  f.u[0] = *(const u32x4*)(row + kb);
  f.u[1] = *(const u32x4*)(row + 16 + kb);
  return f.v;
}
__device__ __forceinline__ v8f wmma16(v16h a, v16h b, v8f c) {
  return __builtin_amdgcn_wmma_f32_16x16x32_f16(false, a, false, b, (short)0, c,
                                                false, false);
}

// ------------------------------- Threefry-2x32 ------------------------------
__device__ __forceinline__ unsigned rotl32(unsigned x, int r) {
  return (x << r) | (x >> (32 - r));
}
__device__ __forceinline__ void threefry2x32(unsigned k0, unsigned k1,
                                             unsigned x0, unsigned x1,
                                             unsigned& o0, unsigned& o1) {
  unsigned ks0 = k0, ks1 = k1, ks2 = 0x1BD11BDAu ^ k0 ^ k1;
  unsigned ks[3] = {ks0, ks1, ks2};
  x0 += ks0; x1 += ks1;
  const int R0[4] = {13, 15, 26, 6};
  const int R1[4] = {17, 29, 16, 24};
#pragma unroll
  for (int g = 0; g < 5; ++g) {
    const int* R = (g & 1) ? R1 : R0;
#pragma unroll
    for (int r = 0; r < 4; ++r) { x0 += x1; x1 = rotl32(x1, R[r]); x1 ^= x0; }
    x0 += ks[(g + 1) % 3];
    x1 += ks[(g + 2) % 3] + (unsigned)(g + 1);
  }
  o0 = x0; o1 = x1;
}
__device__ __forceinline__ float u01(unsigned bits) {
  return __uint_as_float((bits >> 9) | 0x3f800000u) - 1.0f;
}

// keys = jax.random.split(key(42), 100): full[t]=o0(t,100+t), full[100+t]=o1
__global__ void k_keys(unsigned* __restrict__ keys) {
  int t = threadIdx.x;
  if (t < 100) {
    unsigned o0, o1;
    threefry2x32(0u, 42u, (unsigned)t, (unsigned)(100 + t), o0, o1);
    keys[t] = o0;
    keys[100 + t] = o1;
  }
}

// Poisson spike generation, matches jax.random.uniform bit recipe.
__global__ __launch_bounds__(256) void k_poisson(const float* __restrict__ in,
                                                 const unsigned* __restrict__ keys,
                                                 int step,
                                                 _Float16* __restrict__ pois) {
  const int H = 200704; // 401408 / 2
  int i = blockIdx.x * 256 + threadIdx.x;
  if (i >= H) return;
  unsigned k0 = keys[2 * step], k1 = keys[2 * step + 1];
  unsigned o0, o1;
  threefry2x32(k0, k1, (unsigned)i, (unsigned)(H + i), o0, o1);
  float r0 = u01(o0), r1 = u01(o1);
  float x0 = in[i], x1 = in[H + i];
  float s0 = (x0 > 0.f) ? 1.f : ((x0 < 0.f) ? -1.f : 0.f);
  float s1 = (x1 > 0.f) ? 1.f : ((x1 < 0.f) ? -1.f : 0.f);
  pois[i]     = (_Float16)((fabsf(x0) * 0.5f > r0) ? s0 : 0.f);
  pois[H + i] = (_Float16)((fabsf(x1) * 0.5f > r1) ? s1 : 0.f);
}

// Weight prep: pad + permute K + split into f16 hi/lo pair.
// mode 0: w1 (20x25),  k = ky*5+kx
// mode 1: w2 (50x500), k = (ky*5+kx)*20 + ic  -> src oc*500 + ic*25 + tap
// mode 2: wf0(200x2450), k = (y*7+x)*50 + oc -> src n*2450 + oc*49 + pix
// mode 3: wf1 (10x200), k = direct
__global__ __launch_bounds__(256) void k_prep_w(const float* __restrict__ src,
                                                _Float16* __restrict__ dh,
                                                _Float16* __restrict__ dl,
                                                int Nreal, int Kreal, int Npad,
                                                int Kpad, int mode) {
  int i = blockIdx.x * 256 + threadIdx.x;
  if (i >= Npad * Kpad) return;
  int n = i / Kpad, k = i % Kpad;
  float v = 0.f;
  if (n < Nreal && k < Kreal) {
    int si;
    if (mode == 0) si = n * 25 + k;
    else if (mode == 1) { int tap = k / 20, ic = k % 20; si = n * 500 + ic * 25 + tap; }
    else if (mode == 2) { int pix = k / 50, oc = k % 50; si = n * 2450 + oc * 49 + pix; }
    else si = n * 200 + k;
    v = src[si];
  }
  _Float16 h = (_Float16)v;
  dh[i] = h;
  dl[i] = (_Float16)(v - (float)h);
}

// conv1 + fire: implicit GEMM, M=512*784, N=32(pad of 20), K=32(pad of 25).
// 8 waves: 4(M) x 2(N); Mtile=64. State NHWC: m1[b][28][28][20], spikes u8.
__global__ __launch_bounds__(256) void k_conv1(const _Float16* __restrict__ pois,
                                               const _Float16* __restrict__ w1h,
                                               const _Float16* __restrict__ w1l,
                                               float* __restrict__ m1,
                                               unsigned char* __restrict__ o1) {
  __shared__ __align__(16) _Float16 lA[64][32];
  __shared__ __align__(16) _Float16 lBh[32][32];
  __shared__ __align__(16) _Float16 lBl[32][32];
  const int tid = threadIdx.x;
  const unsigned lane = tid & 31u;
  const int wave = tid >> 5;
  const int wm = wave >> 1, wn = wave & 1;
  const int mBase = blockIdx.x * 64;

  if (tid < 128)      ((u32x4*)lBh)[tid]       = ((const u32x4*)w1h)[tid];
  else                ((u32x4*)lBl)[tid - 128] = ((const u32x4*)w1l)[tid - 128];

#pragma unroll
  for (int j = 0; j < 8; ++j) {
    int e = tid * 8 + j;
    int ml = e >> 5, k = e & 31;
    int m = mBase + ml;
    int b = m / 784, p = m % 784;
    int y = p / 28, x = p % 28;
    float v = 0.f;
    if (k < 25) {
      int ky = k / 5, kx = k % 5;
      int yy = y + ky - 2, xx = x + kx - 2;
      if ((unsigned)yy < 28u && (unsigned)xx < 28u)
        v = (float)pois[b * 784 + yy * 28 + xx];
    }
    lA[ml][k] = (_Float16)v;
  }
  __syncthreads();

  v16h a  = load_a_frag(&lA[wm * 16 + (int)(lane & 15u)][0], lane);
  v16h bh = load_b_frag(&lBh[wn * 16 + (int)(lane & 15u)][0], lane);
  v16h bl = load_b_frag(&lBl[wn * 16 + (int)(lane & 15u)][0], lane);
  v8f acc = {};
  acc = wmma16(a, bh, acc);
  acc = wmma16(a, bl, acc);

  int oc = wn * 16 + (int)(lane & 15u);
  if (oc < 20) {
    int rowoff = (lane & 16u) ? 8 : 0;
#pragma unroll
    for (int r = 0; r < 8; ++r) {
      int m = mBase + wm * 16 + rowoff + r;
      int idx = m * 20 + oc;
      float mem = m1[idx] + acc[r];
      bool sp = mem > 1.0f;
      m1[idx] = sp ? 0.f : mem;
      o1[idx] = sp ? (unsigned char)1 : (unsigned char)0;
    }
  }
}

// 2x2 avg-pool + fire (threshold th). NHWC u8 spikes in/out, f32 membrane.
__global__ __launch_bounds__(256) void k_pool(const unsigned char* __restrict__ in,
                                              float* __restrict__ mem,
                                              unsigned char* __restrict__ out,
                                              int C, int HWo, int n_total,
                                              float th) {
  int i = blockIdx.x * 256 + threadIdx.x;
  if (i >= n_total) return;
  int c = i % C;
  int t = i / C;
  int x = t % HWo;
  int t2 = t / HWo;
  int y = t2 % HWo;
  int b = t2 / HWo;
  int Wi = HWo * 2;
  int base = ((b * Wi + 2 * y) * Wi + 2 * x) * C + c;
  float s = (float)(in[base] + in[base + C] + in[base + Wi * C] + in[base + Wi * C + C]);
  float m = mem[i] + s * 0.25f;
  bool sp = m > th;
  mem[i] = sp ? 0.f : m;
  out[i] = sp ? (unsigned char)1 : (unsigned char)0;
}

// conv2 + fire: implicit GEMM, M=512*196, N=64(pad of 50), K=512(pad of 500).
// 8 waves: 2(M) x 4(N); Mtile=32. K permuted as (tap)*20+ic for contiguous im2col.
__global__ __launch_bounds__(256) void k_conv2(const unsigned char* __restrict__ op1,
                                               const _Float16* __restrict__ w2h,
                                               const _Float16* __restrict__ w2l,
                                               float* __restrict__ m2,
                                               unsigned char* __restrict__ o2) {
  __shared__ __align__(16) _Float16 lA[32][32];
  __shared__ __align__(16) _Float16 lBh[64][32];
  __shared__ __align__(16) _Float16 lBl[64][32];
  const int tid = threadIdx.x;
  const unsigned lane = tid & 31u;
  const int wave = tid >> 5;
  const int wm = wave & 1, wn = wave >> 1;
  const int mBase = blockIdx.x * 32;
  const int rowB = tid >> 2, offB = (tid & 3) * 8;

  v8f acc = {};
  for (int kc = 0; kc < 16; ++kc) {
    __syncthreads();
    if (kc + 1 < 16) __builtin_prefetch(w2h + rowB * 512 + (kc + 1) * 32 + offB, 0, 3);
    *(u32x4*)&lBh[rowB][offB] = *(const u32x4*)(w2h + rowB * 512 + kc * 32 + offB);
    *(u32x4*)&lBl[rowB][offB] = *(const u32x4*)(w2l + rowB * 512 + kc * 32 + offB);
#pragma unroll
    for (int j = 0; j < 4; ++j) {
      int e = tid * 4 + j;
      int ml = e >> 5, k = e & 31;
      int m = mBase + ml;
      int b = m / 196, p = m % 196;
      int y = p / 14, x = p % 14;
      int kk = kc * 32 + k;
      float v = 0.f;
      if (kk < 500) {
        int tap = kk / 20, ic = kk % 20;
        int ky = tap / 5, kx = tap % 5;
        int yy = y + ky - 2, xx = x + kx - 2;
        if ((unsigned)yy < 14u && (unsigned)xx < 14u)
          v = (float)op1[((b * 14 + yy) * 14 + xx) * 20 + ic];
      }
      lA[ml][k] = (_Float16)v;
    }
    __syncthreads();
    v16h a  = load_a_frag(&lA[wm * 16 + (int)(lane & 15u)][0], lane);
    v16h bh = load_b_frag(&lBh[wn * 16 + (int)(lane & 15u)][0], lane);
    v16h bl = load_b_frag(&lBl[wn * 16 + (int)(lane & 15u)][0], lane);
    acc = wmma16(a, bh, acc);
    acc = wmma16(a, bl, acc);
  }

  int oc = wn * 16 + (int)(lane & 15u);
  if (oc < 50) {
    int rowoff = (lane & 16u) ? 8 : 0;
#pragma unroll
    for (int r = 0; r < 8; ++r) {
      int m = mBase + wm * 16 + rowoff + r;
      int idx = m * 50 + oc;
      float mem = m2[idx] + acc[r];
      bool sp = mem > 1.0f;
      m2[idx] = sp ? 0.f : mem;
      o2[idx] = sp ? (unsigned char)1 : (unsigned char)0;
    }
  }
}

// fc0 + fire + Tf0 accumulate: GEMM M=512, N=256(pad of 200), K=2464(pad of 2450).
// K is permuted (pix*50+oc) at prep time so A == flat NHWC op2 spikes.
__global__ __launch_bounds__(256) void k_fc0(const unsigned char* __restrict__ op2,
                                             const _Float16* __restrict__ wf0h,
                                             const _Float16* __restrict__ wf0l,
                                             float* __restrict__ mf0,
                                             float* __restrict__ tf0) {
  __shared__ __align__(16) _Float16 lA[32][32];
  __shared__ __align__(16) _Float16 lBh[64][32];
  __shared__ __align__(16) _Float16 lBl[64][32];
  const int tid = threadIdx.x;
  const unsigned lane = tid & 31u;
  const int wave = tid >> 5;
  const int wm = wave & 1, wn = wave >> 1;
  const int mBase = blockIdx.x * 32;
  const int nBase = blockIdx.y * 64;
  const int rowB = tid >> 2, offB = (tid & 3) * 8;

  v8f acc = {};
  for (int kc = 0; kc < 77; ++kc) {
    __syncthreads();
    if (kc + 1 < 77)
      __builtin_prefetch(wf0h + (nBase + rowB) * 2464 + (kc + 1) * 32 + offB, 0, 3);
    *(u32x4*)&lBh[rowB][offB] = *(const u32x4*)(wf0h + (nBase + rowB) * 2464 + kc * 32 + offB);
    *(u32x4*)&lBl[rowB][offB] = *(const u32x4*)(wf0l + (nBase + rowB) * 2464 + kc * 32 + offB);
#pragma unroll
    for (int j = 0; j < 4; ++j) {
      int e = tid * 4 + j;
      int ml = e >> 5, k = e & 31;
      int b = mBase + ml;
      int kk = kc * 32 + k;
      float v = (kk < 2450) ? (float)op2[b * 2450 + kk] : 0.f;
      lA[ml][k] = (_Float16)v;
    }
    __syncthreads();
    v16h a  = load_a_frag(&lA[wm * 16 + (int)(lane & 15u)][0], lane);
    v16h bh = load_b_frag(&lBh[wn * 16 + (int)(lane & 15u)][0], lane);
    v16h bl = load_b_frag(&lBl[wn * 16 + (int)(lane & 15u)][0], lane);
    acc = wmma16(a, bh, acc);
    acc = wmma16(a, bl, acc);
  }

  int n = nBase + wn * 16 + (int)(lane & 15u);
  if (n < 200) {
    int rowoff = (lane & 16u) ? 8 : 0;
#pragma unroll
    for (int r = 0; r < 8; ++r) {
      int b = mBase + wm * 16 + rowoff + r;
      int idx = b * 200 + n;
      float mem = mf0[idx] + acc[r];
      bool sp = mem > 1.0f;
      mf0[idx] = sp ? 0.f : mem;
      if (sp) tf0[idx] += 1.0f;
    }
  }
}

// Final: out = (Tf0 @ wf1^T) / 100. M=512, N=16(pad of 10), K=224(pad of 200).
// 8 waves stacked on M; Mtile=128.
__global__ __launch_bounds__(256) void k_final(const float* __restrict__ tf0,
                                               const _Float16* __restrict__ wf1h,
                                               const _Float16* __restrict__ wf1l,
                                               float* __restrict__ out) {
  __shared__ __align__(16) _Float16 lA[128][32];
  __shared__ __align__(16) _Float16 lBh[16][32];
  __shared__ __align__(16) _Float16 lBl[16][32];
  const int tid = threadIdx.x;
  const unsigned lane = tid & 31u;
  const int wm = tid >> 5;
  const int mBase = blockIdx.x * 128;

  v8f acc = {};
  for (int kc = 0; kc < 7; ++kc) {
    __syncthreads();
    if (tid < 64) {
      int r = tid >> 2, o = (tid & 3) * 8;
      *(u32x4*)&lBh[r][o] = *(const u32x4*)(wf1h + r * 224 + kc * 32 + o);
    } else if (tid < 128) {
      int t = tid - 64;
      int r = t >> 2, o = (t & 3) * 8;
      *(u32x4*)&lBl[r][o] = *(const u32x4*)(wf1l + r * 224 + kc * 32 + o);
    }
#pragma unroll
    for (int j = 0; j < 16; ++j) {
      int e = tid * 16 + j;
      int ml = e >> 5, k = e & 31;
      int b = mBase + ml;
      int kk = kc * 32 + k;
      lA[ml][k] = (_Float16)((kk < 200) ? tf0[b * 200 + kk] : 0.f);
    }
    __syncthreads();
    v16h a  = load_a_frag(&lA[wm * 16 + (int)(lane & 15u)][0], lane);
    v16h bh = load_b_frag(&lBh[(int)(lane & 15u)][0], lane);
    v16h bl = load_b_frag(&lBl[(int)(lane & 15u)][0], lane);
    acc = wmma16(a, bh, acc);
    acc = wmma16(a, bl, acc);
  }

  int col = (int)(lane & 15u);
  if (col < 10) {
    int rowoff = (lane & 16u) ? 8 : 0;
#pragma unroll
    for (int r = 0; r < 8; ++r) {
      int b = mBase + wm * 16 + rowoff + r;
      out[b * 10 + col] = acc[r] / 100.0f;
    }
  }
}

// ---------------------------------------------------------------------------
extern "C" void kernel_launch(void* const* d_in, const int* in_sizes, int n_in,
                              void* d_out, int out_size, void* d_ws, size_t ws_size,
                              hipStream_t stream) {
  (void)in_sizes; (void)n_in; (void)out_size; (void)ws_size;
  const float* inp  = (const float*)d_in[0]; // (512,1,28,28)
  const float* w1   = (const float*)d_in[1]; // (20,1,5,5)
  const float* w2   = (const float*)d_in[2]; // (50,20,5,5)
  const float* wf0  = (const float*)d_in[3]; // (200,2450)
  const float* wf1  = (const float*)d_in[4]; // (10,200)
  float* out = (float*)d_out;                // (512,10)

  char* ws = (char*)d_ws;
  size_t off = 0;
  auto alloc = [&](size_t bytes) -> char* {
    char* p = ws + off;
    off = (off + bytes + 255) & ~(size_t)255;
    return p;
  };

  unsigned*      keys = (unsigned*)alloc(200 * 4);
  _Float16*      w1h  = (_Float16*)alloc(32 * 32 * 2);
  _Float16*      w1l  = (_Float16*)alloc(32 * 32 * 2);
  _Float16*      w2h  = (_Float16*)alloc(64 * 512 * 2);
  _Float16*      w2l  = (_Float16*)alloc(64 * 512 * 2);
  _Float16*      wf0h = (_Float16*)alloc((size_t)256 * 2464 * 2);
  _Float16*      wf0l = (_Float16*)alloc((size_t)256 * 2464 * 2);
  _Float16*      wf1h = (_Float16*)alloc(16 * 224 * 2);
  _Float16*      wf1l = (_Float16*)alloc(16 * 224 * 2);
  _Float16*      pois = (_Float16*)alloc((size_t)512 * 784 * 2);
  float*         m1   = (float*)alloc((size_t)512 * 784 * 20 * 4);
  unsigned char* o1   = (unsigned char*)alloc((size_t)512 * 784 * 20);
  float*         m1s  = (float*)alloc((size_t)512 * 196 * 20 * 4);
  unsigned char* op1  = (unsigned char*)alloc((size_t)512 * 196 * 20);
  float*         m2   = (float*)alloc((size_t)512 * 196 * 50 * 4);
  unsigned char* o2   = (unsigned char*)alloc((size_t)512 * 196 * 50);
  float*         m2s  = (float*)alloc((size_t)512 * 49 * 50 * 4);
  unsigned char* op2  = (unsigned char*)alloc((size_t)512 * 49 * 50);
  float*         mf0  = (float*)alloc((size_t)512 * 200 * 4);
  float*         tf0  = (float*)alloc((size_t)512 * 200 * 4);
  // total workspace ~86 MB

  // zero persistent state every call (deterministic; capturable memset nodes)
  hipMemsetAsync(m1,  0, (size_t)512 * 784 * 20 * 4, stream);
  hipMemsetAsync(m1s, 0, (size_t)512 * 196 * 20 * 4, stream);
  hipMemsetAsync(m2,  0, (size_t)512 * 196 * 50 * 4, stream);
  hipMemsetAsync(m2s, 0, (size_t)512 * 49 * 50 * 4, stream);
  hipMemsetAsync(mf0, 0, (size_t)512 * 200 * 4, stream);
  hipMemsetAsync(tf0, 0, (size_t)512 * 200 * 4, stream);

  k_keys<<<1, 128, 0, stream>>>(keys);
  k_prep_w<<<(32 * 32 + 255) / 256, 256, 0, stream>>>(w1, w1h, w1l, 20, 25, 32, 32, 0);
  k_prep_w<<<(64 * 512 + 255) / 256, 256, 0, stream>>>(w2, w2h, w2l, 50, 500, 64, 512, 1);
  k_prep_w<<<(256 * 2464 + 255) / 256, 256, 0, stream>>>(wf0, wf0h, wf0l, 200, 2450, 256, 2464, 2);
  k_prep_w<<<(16 * 224 + 255) / 256, 256, 0, stream>>>(wf1, wf1h, wf1l, 10, 200, 16, 224, 3);

  for (int t = 0; t < 100; ++t) {
    k_poisson<<<784, 256, 0, stream>>>(inp, keys, t, pois);
    k_conv1<<<6272, 256, 0, stream>>>(pois, w1h, w1l, m1, o1);
    k_pool<<<2007040 / 256, 256, 0, stream>>>(o1, m1s, op1, 20, 14, 2007040, 0.75f);
    k_conv2<<<3136, 256, 0, stream>>>(op1, w2h, w2l, m2, o2);
    k_pool<<<1254400 / 256, 256, 0, stream>>>(o2, m2s, op2, 50, 7, 1254400, 0.75f);
    k_fc0<<<dim3(16, 4), 256, 0, stream>>>(op2, wf0h, wf0l, mf0, tf0);
  }

  k_final<<<4, 256, 0, stream>>>(tf0, wf1h, wf1l, out);
}
